// PreLoraAttention_54314156425797
// MI455X (gfx1250) — matrix operand order repaired and checked by reading, not compiled, not run on hardware
//
#include <hip/hip_runtime.h>

// MI455X (gfx1250) LoRA attention, bf16 WMMA end-to-end.
// LoRA folded into W_eff = W + 2.0 * B@A, so the problem is 4 GEMMs + flash
// attention. GEMMs: 128x128 block tiles staged in LDS via
// global_load_async_to_lds_b128 (ASYNCcnt, double buffered), 32x64 wave
// subtiles on v_wmma_f32_16x16x32_bf16.

namespace {

constexpr int Bb = 4, S = 4096, Pp = 64, H = 1024, R = 8;
constexpr int Tt = Pp + S;       // 4160 keys/values after cache concat
constexpr int BS = Bb * S;       // 16384 query rows
constexpr float LORA_SCALE  = 16.0f / (float)R;  // 2.0
constexpr float SCORE_SCALE = 0.03125f;          // 1/sqrt(1024)

typedef __attribute__((ext_vector_type(16))) __bf16 bf16x16;
typedef __attribute__((ext_vector_type(8)))  float  f32x8;

union FragB {
  bf16x16 v;
  uint4   q[2];
};

__device__ __forceinline__ unsigned short f2bf(float f) {
  unsigned int u = __float_as_uint(f);
  u += 0x7FFFu + ((u >> 16) & 1u);   // round-to-nearest-even
  return (unsigned short)(u >> 16);
}

__device__ __forceinline__ f32x8 wmma_bf16(const FragB& a, const FragB& b, f32x8 c) {
  return __builtin_amdgcn_wmma_f32_16x16x32_bf16(false, a.v, false, b.v,
                                                 (short)0, c, false, false);
}

// ---- CDNA5 async memory->LDS copy (tracked by ASYNCcnt) --------------------
__device__ __forceinline__ void async_b128(unsigned lds_byte_addr,
                                           const unsigned short* gptr) {
  asm volatile("global_load_async_to_lds_b128 %0, %1, off"
               :: "v"(lds_byte_addr), "v"(gptr) : "memory");
}

__device__ __forceinline__ void wait_async_le4() {
#if __has_builtin(__builtin_amdgcn_s_wait_asynccnt)
  __builtin_amdgcn_s_wait_asynccnt(4);
#else
  asm volatile("s_wait_asynccnt 0x4" ::: "memory");
#endif
}

__device__ __forceinline__ void wait_async_0() {
#if __has_builtin(__builtin_amdgcn_s_wait_asynccnt)
  __builtin_amdgcn_s_wait_asynccnt(0);
#else
  asm volatile("s_wait_asynccnt 0x0" ::: "memory");
#endif
}

__device__ __forceinline__ unsigned lds_addr(const void* p) {
  return (unsigned)(unsigned long long)p;
}

// ---- W_eff[n,h] = W[n,h] + 2.0 * sum_r B[n,r]*A[r,h]  ->  bf16 --------------
__global__ void k_prep_weight(const float* __restrict__ W, const float* __restrict__ A,
                              const float* __restrict__ Bm, unsigned short* __restrict__ weff) {
  int idx = blockIdx.x * 256 + threadIdx.x;
  if (idx >= H * H) return;
  int n = idx >> 10, h = idx & (H - 1);
  float acc = W[idx];
#pragma unroll
  for (int r = 0; r < R; ++r) acc += LORA_SCALE * Bm[n * R + r] * A[r * H + h];
  weff[idx] = f2bf(acc);
}

// ---- fp32 -> bf16 converters -----------------------------------------------
__global__ void k_conv_x(const float* __restrict__ x, unsigned short* __restrict__ xbf) {
  size_t idx = (size_t)blockIdx.x * 256 + threadIdx.x;
  if (idx < (size_t)BS * H) xbf[idx] = f2bf(x[idx]);
}

__global__ void k_conv_pastk(const float* __restrict__ pk, unsigned short* __restrict__ kfull) {
  int idx = blockIdx.x * 256 + threadIdx.x;   // Bb*Pp*H
  if (idx >= Bb * Pp * H) return;
  int b = idx / (Pp * H), t = (idx / H) % Pp, h = idx & (H - 1);
  kfull[((size_t)b * Tt + t) * H + h] = f2bf(pk[idx]);
}

__global__ void k_conv_pastv(const float* __restrict__ pv, unsigned short* __restrict__ vT) {
  int idx = blockIdx.x * 256 + threadIdx.x;   // Bb*Pp*H ; V stored transposed [B,H,T]
  if (idx >= Bb * Pp * H) return;
  int b = idx / (Pp * H), t = (idx / H) % Pp, h = idx & (H - 1);
  vT[((size_t)b * H + h) * Tt + t] = f2bf(pv[idx]);
}

// ---- unified projection GEMM: out = act @ W_eff[proj]^T + bias -------------
// proj0 = blockIdx-derived projection + projBase (0 for q/k/v grid, 3 for o).
__global__ __launch_bounds__(256) void k_gemm(
    int projBase,
    const unsigned short* __restrict__ act,   // activations (xbf or ctx)
    const unsigned short* __restrict__ weff,  // [4][H][H] bf16
    const float* __restrict__ bq, const float* __restrict__ bk,
    const float* __restrict__ bv, const float* __restrict__ bo,
    unsigned short* __restrict__ qbf, unsigned short* __restrict__ kfull,
    unsigned short* __restrict__ vT, float* __restrict__ out) {
  __shared__ unsigned short s_x[2][128 * 32];   // 2 x 8KB
  __shared__ unsigned short s_w[2][128 * 32];   // 2 x 8KB

  const int tid = threadIdx.x;
  const int wv = tid >> 5, lane = tid & 31;
  const int half = lane >> 4, col = lane & 15;
  const int ms = wv & 3, ns = wv >> 2;          // 4x2 wave grid over 128x128

  const int bid = blockIdx.x;
  const int proj = projBase + (bid >> 10);      // 1024 blocks per projection
  const int rem = bid & 1023;
  const int mblk = rem >> 3, nblk = rem & 7;

  const unsigned short* xbase = act + (size_t)(mblk * 128) * H;
  const unsigned short* wbase = weff + (size_t)proj * H * H + (size_t)(nblk * 128) * H;

  // this thread's two 16B chunks of each 8KB tile
  const int chunk0 = tid * 2;
  const int row0 = chunk0 >> 2, quad0 = chunk0 & 3;
  const int row1 = (chunk0 + 1) >> 2, quad1 = (chunk0 + 1) & 3;

  const f32x8 z = {0.f, 0.f, 0.f, 0.f, 0.f, 0.f, 0.f, 0.f};
  f32x8 acc[2][4];
#pragma unroll
  for (int mt = 0; mt < 2; ++mt)
#pragma unroll
    for (int nt = 0; nt < 4; ++nt) acc[mt][nt] = z;

  constexpr int NSTEP = H / 32;   // 32

  // prologue: stage step 0 into buffer 0
  async_b128(lds_addr(&s_x[0][chunk0 * 8]),     xbase + (size_t)row0 * H + quad0 * 8);
  async_b128(lds_addr(&s_x[0][chunk0 * 8 + 8]), xbase + (size_t)row1 * H + quad1 * 8);
  async_b128(lds_addr(&s_w[0][chunk0 * 8]),     wbase + (size_t)row0 * H + quad0 * 8);
  async_b128(lds_addr(&s_w[0][chunk0 * 8 + 8]), wbase + (size_t)row1 * H + quad1 * 8);

  for (int step = 0; step < NSTEP; ++step) {
    const int cur = step & 1, nxt = cur ^ 1;
    if (step + 1 < NSTEP) {
      const int kb = (step + 1) * 32;
      async_b128(lds_addr(&s_x[nxt][chunk0 * 8]),     xbase + (size_t)row0 * H + kb + quad0 * 8);
      async_b128(lds_addr(&s_x[nxt][chunk0 * 8 + 8]), xbase + (size_t)row1 * H + kb + quad1 * 8);
      async_b128(lds_addr(&s_w[nxt][chunk0 * 8]),     wbase + (size_t)row0 * H + kb + quad0 * 8);
      async_b128(lds_addr(&s_w[nxt][chunk0 * 8 + 8]), wbase + (size_t)row1 * H + kb + quad1 * 8);
      wait_async_le4();   // async completes in-order: current tile has landed
    } else {
      wait_async_0();
    }
    __syncthreads();      // publish current LDS tile to all waves

    FragB a[2];
#pragma unroll
    for (int mt = 0; mt < 2; ++mt) {
      const unsigned short* xr = &s_x[cur][(ms * 32 + mt * 16 + col) * 32];
      a[mt].q[0] = *(const uint4*)(xr + half * 8);
      a[mt].q[1] = *(const uint4*)(xr + 16 + half * 8);
    }
#pragma unroll
    for (int nt = 0; nt < 4; ++nt) {
      const unsigned short* wr = &s_w[cur][(ns * 64 + nt * 16 + col) * 32];
      FragB bfr;
      bfr.q[0] = *(const uint4*)(wr + half * 8);
      bfr.q[1] = *(const uint4*)(wr + 16 + half * 8);
      acc[0][nt] = wmma_bf16(a[0], bfr, acc[0][nt]);
      acc[1][nt] = wmma_bf16(a[1], bfr, acc[1][nt]);
    }
    __syncthreads();      // done reading buffer `cur` before refill
  }

  const float* bias = (proj == 0) ? bq : (proj == 1) ? bk : (proj == 2) ? bv : bo;
#pragma unroll
  for (int nt = 0; nt < 4; ++nt) {
    const int n = nblk * 128 + ns * 64 + nt * 16 + col;
    const float bn = bias[n];
#pragma unroll
    for (int mt = 0; mt < 2; ++mt) {
#pragma unroll
      for (int r = 0; r < 8; ++r) {
        const int m = mblk * 128 + ms * 32 + mt * 16 + half * 8 + r;
        const float val = acc[mt][nt][r] + bn;
        const int bi = m >> 12, s = m & (S - 1);
        if (proj == 0)      qbf[(size_t)m * H + n] = f2bf(val);
        else if (proj == 1) kfull[((size_t)bi * Tt + Pp + s) * H + n] = f2bf(val);
        else if (proj == 2) vT[((size_t)bi * H + n) * Tt + Pp + s] = f2bf(val);
        else                out[(size_t)m * H + n] = val;
      }
    }
  }
}

// ---- cooperative flash attention -------------------------------------------
__global__ __launch_bounds__(256) void k_attn(
    const unsigned short* __restrict__ qbf, const unsigned short* __restrict__ kfull,
    const unsigned short* __restrict__ vT, unsigned short* __restrict__ ctxbf) {
  __shared__ float s_part[8][16][16];
  __shared__ unsigned short s_P[16][32];

  const int wv = threadIdx.x >> 5, lane = threadIdx.x & 31;
  const int half = lane >> 4, col = lane & 15;
  const int b = blockIdx.x >> 8;
  const int mt = blockIdx.x & 255;
  const int grow = b * S + mt * 16;

  FragB qa[4];
  {
    const unsigned short* qrow = qbf + (size_t)(grow + col) * H + wv * 128;
#pragma unroll
    for (int j = 0; j < 4; ++j) {
      qa[j].q[0] = *(const uint4*)(qrow + j * 32 + half * 8);
      qa[j].q[1] = *(const uint4*)(qrow + j * 32 + 16 + half * 8);
    }
  }

  const f32x8 z = {0.f, 0.f, 0.f, 0.f, 0.f, 0.f, 0.f, 0.f};
  f32x8 acc[8];
#pragma unroll
  for (int t = 0; t < 8; ++t) acc[t] = z;
  float m_run[8], l_run[8];
#pragma unroll
  for (int r = 0; r < 8; ++r) { m_run[r] = -1e30f; l_run[r] = 0.f; }

  const unsigned short* kbbase = kfull + (size_t)b * Tt * H;
  const unsigned short* vbase  = vT + (size_t)b * H * Tt;

  for (int c = 0; c < Tt / 32; ++c) {
    const int kbase = c * 32;
    f32x8 st[2];
#pragma unroll
    for (int sub = 0; sub < 2; ++sub) {
      f32x8 sp = z;
      const int kr = kbase + sub * 16 + col;
      const unsigned short* krow = kbbase + (size_t)kr * H + wv * 128 + half * 8;
#pragma unroll
      for (int j = 0; j < 4; ++j) {
        FragB kf;
        kf.q[0] = *(const uint4*)(krow + j * 32);
        kf.q[1] = *(const uint4*)(krow + j * 32 + 16);
        sp = wmma_bf16(qa[j], kf, sp);
      }
#pragma unroll
      for (int r = 0; r < 8; ++r) s_part[wv][half * 8 + r][col] = sp[r];
      __syncthreads();
      f32x8 full = z;
#pragma unroll
      for (int w = 0; w < 8; ++w)
#pragma unroll
        for (int r = 0; r < 8; ++r) full[r] += s_part[w][half * 8 + r][col];
      __syncthreads();
#pragma unroll
      for (int r = 0; r < 8; ++r) full[r] *= SCORE_SCALE;
      st[sub] = full;
    }

    float pf0[8], pf1[8];
#pragma unroll
    for (int r = 0; r < 8; ++r) {
      float v = fmaxf(st[0][r], st[1][r]);
#pragma unroll
      for (int msk = 1; msk < 16; msk <<= 1) v = fmaxf(v, __shfl_xor(v, msk, 32));
      const float mnew = fmaxf(m_run[r], v);
      const float sc = __expf(m_run[r] - mnew);
      const float p0 = __expf(st[0][r] - mnew);
      const float p1 = __expf(st[1][r] - mnew);
      float ps = p0 + p1;
#pragma unroll
      for (int msk = 1; msk < 16; msk <<= 1) ps += __shfl_xor(ps, msk, 32);
      l_run[r] = l_run[r] * sc + ps;
      m_run[r] = mnew;
      pf0[r] = p0; pf1[r] = p1;
#pragma unroll
      for (int t = 0; t < 8; ++t) acc[t][r] *= sc;
    }
    if (wv == 0) {
#pragma unroll
      for (int r = 0; r < 8; ++r) {
        s_P[half * 8 + r][col]      = f2bf(pf0[r]);
        s_P[half * 8 + r][16 + col] = f2bf(pf1[r]);
      }
    }
    __syncthreads();

    FragB pa;
    pa.q[0] = *(const uint4*)&s_P[col][half * 8];
    pa.q[1] = *(const uint4*)&s_P[col][16 + half * 8];

    if (c + 1 < Tt / 32)
      __builtin_prefetch(kbbase + (size_t)(kbase + 32 + col) * H + wv * 128, 0, 1);

#pragma unroll
    for (int t = 0; t < 8; ++t) {
      const int n = wv * 128 + t * 16 + col;
      const unsigned short* vrow = vbase + (size_t)n * Tt + kbase + half * 8;
      FragB vf;
      vf.q[0] = *(const uint4*)(vrow);
      vf.q[1] = *(const uint4*)(vrow + 16);
      acc[t] = wmma_bf16(pa, vf, acc[t]);
    }
    __syncthreads();
  }

#pragma unroll
  for (int t = 0; t < 8; ++t) {
    const int n = wv * 128 + t * 16 + col;
#pragma unroll
    for (int r = 0; r < 8; ++r) {
      const float val = acc[t][r] / l_run[r];
      ctxbf[(size_t)(grow + half * 8 + r) * H + n] = f2bf(val);
    }
  }
}

}  // namespace

extern "C" void kernel_launch(void* const* d_in, const int* in_sizes, int n_in,
                              void* d_out, int out_size, void* d_ws, size_t ws_size,
                              hipStream_t stream) {
  (void)in_sizes; (void)n_in; (void)out_size; (void)ws_size;
  const float* x  = (const float*)d_in[0];
  const float* pk = (const float*)d_in[1];
  const float* pv = (const float*)d_in[2];
  const float* Wq = (const float*)d_in[3];  const float* bq = (const float*)d_in[4];
  const float* Aq = (const float*)d_in[5];  const float* Bq = (const float*)d_in[6];
  const float* Wk = (const float*)d_in[7];  const float* bk = (const float*)d_in[8];
  const float* Ak = (const float*)d_in[9];  const float* Bk = (const float*)d_in[10];
  const float* Wv = (const float*)d_in[11]; const float* bv = (const float*)d_in[12];
  const float* Av = (const float*)d_in[13]; const float* Bv = (const float*)d_in[14];
  const float* Wo = (const float*)d_in[15]; const float* bo = (const float*)d_in[16];
  const float* Ao = (const float*)d_in[17]; const float* Bo = (const float*)d_in[18];

  unsigned short* weff  = (unsigned short*)d_ws;              // 4*H*H
  unsigned short* xbf   = weff  + (size_t)4 * H * H;          // BS*H
  unsigned short* qbf   = xbf   + (size_t)BS * H;             // BS*H
  unsigned short* kfull = qbf   + (size_t)BS * H;             // Bb*Tt*H
  unsigned short* vT    = kfull + (size_t)Bb * Tt * H;        // Bb*H*Tt
  unsigned short* ctx   = vT    + (size_t)Bb * Tt * H;        // BS*H

  const int wblk = (H * H + 255) / 256;
  k_prep_weight<<<wblk, 256, 0, stream>>>(Wq, Aq, Bq, weff + (size_t)0 * H * H);
  k_prep_weight<<<wblk, 256, 0, stream>>>(Wk, Ak, Bk, weff + (size_t)1 * H * H);
  k_prep_weight<<<wblk, 256, 0, stream>>>(Wv, Av, Bv, weff + (size_t)2 * H * H);
  k_prep_weight<<<wblk, 256, 0, stream>>>(Wo, Ao, Bo, weff + (size_t)3 * H * H);

  k_conv_x<<<(int)(((size_t)BS * H + 255) / 256), 256, 0, stream>>>(x, xbf);
  k_conv_pastk<<<(Bb * Pp * H + 255) / 256, 256, 0, stream>>>(pk, kfull);
  k_conv_pastv<<<(Bb * Pp * H + 255) / 256, 256, 0, stream>>>(pv, vT);

  // q/k/v projections: 3 * (16384/128) * (1024/128) = 3072 blocks
  k_gemm<<<3 * (BS / 128) * (H / 128), 256, 0, stream>>>(
      0, xbf, weff, bq, bk, bv, bo, qbf, kfull, vT, (float*)d_out);

  k_attn<<<Bb * (S / 16), 256, 0, stream>>>(qbf, kfull, vT, ctx);

  // o projection: 1024 blocks, projBase = 3, activations = ctx, fp32 output
  k_gemm<<<(BS / 128) * (H / 128), 256, 0, stream>>>(
      3, ctx, weff, bq, bk, bv, bo, qbf, kfull, vT, (float*)d_out);
}
